// RNNSeq2Seq_35175782154688
// MI455X (gfx1250) — matrix operand order repaired and checked by reading, not compiled, not run on hardware
//
#include <hip/hip_runtime.h>
#include <stdint.h>
#include <stddef.h>

// ---------------------------------------------------------------------------
// MI455X (gfx1250) seq2seq BiLSTM. wave32, WMMA bf16 (16x16x32) fp32 accum,
// TDM (tensor_load_to_lds) double-buffered GEMM staging with HW LDS padding.
// ---------------------------------------------------------------------------

typedef __attribute__((ext_vector_type(16))) __bf16 v16bf;
typedef __attribute__((ext_vector_type(8)))  float  v8f;
typedef __attribute__((ext_vector_type(4)))  unsigned int u32x4;
typedef __attribute__((ext_vector_type(8)))  int          i32x8;
typedef __attribute__((ext_vector_type(4)))  int          i32x4;

#if __has_builtin(__builtin_amdgcn_tensor_load_to_lds) && \
    __has_builtin(__builtin_amdgcn_s_wait_tensorcnt)
#define USE_TDM 1
#else
#define USE_TDM 0
#endif

struct __attribute__((aligned(16))) U4 { unsigned int x, y, z, w; };

union Fr16 {
  U4    u[2];
  v16bf v;
};

__device__ __forceinline__ unsigned short f2bf(float f) {
  unsigned int u = __float_as_uint(f);
  u += 0x7FFFu + ((u >> 16) & 1u);        // round-to-nearest-even
  return (unsigned short)(u >> 16);
}
__device__ __forceinline__ float bf2f(unsigned short h) {
  return __uint_as_float(((unsigned int)h) << 16);
}
__device__ __forceinline__ float sigf(float x) { return 1.0f / (1.0f + __expf(-x)); }

__device__ __forceinline__ v8f wmma_bf16(const Fr16& a, const Fr16& b, v8f c) {
  return __builtin_amdgcn_wmma_f32_16x16x32_bf16(
      /*neg_a=*/false, a.v, /*neg_b=*/false, b.v,
      /*c_mod=*/(short)0, c, /*reuse_a=*/false, /*reuse_b=*/false);
}

#if USE_TDM
// TDM: load a 128-row x 64-byte (32 bf16) tile into LDS, with 16B pad after
// every 64B row (-> 40-u16 LDS row stride). Element size 8B, tile_dim0 = 8.
// Rows >= valid_rows read as zero via tensor_dim1 OOB (used for dense M=64).
__device__ __forceinline__ void tdm_load_tile_128x32(
    const unsigned short* gbase, long elem_off, int row_stride_elems,
    int valid_rows, unsigned lds_addr) {
  const unsigned long long ga =
      (unsigned long long)gbase + (unsigned long long)elem_off * 2ull;
  u32x4 g0;
  g0[0] = 1u;                                         // count=1 (user D#)
  g0[1] = lds_addr;                                   // LDS byte address
  g0[2] = (unsigned)(ga & 0xFFFFFFFFull);             // global_addr[31:0]
  g0[3] = (unsigned)((ga >> 32) & 0x01FFFFFFull) | 0x80000000u; // [56:32]|type=2
  i32x8 g1;
  g1[0] = (3 << 16) | (1 << 20) | (3 << 22) | (3 << 25); // 8B elems, pad 4dw/16dw
  g1[1] = (int)(8u << 16);                               // tensor_dim0 = 8 units
  const unsigned vr = (unsigned)valid_rows;
  g1[2] = (int)((vr & 0xFFFFu) << 16);                   // tensor_dim1 lo
  g1[3] = (int)((vr >> 16) | (8u << 16));                // tensor_dim1 hi | tile_dim0=8
  g1[4] = 128;                                           // tile_dim1=128 rows
  g1[5] = row_stride_elems >> 2;                         // dim0 stride (8B units)
  g1[6] = 0;
  g1[7] = 0;
  const i32x4 z4 = {0, 0, 0, 0};
  const i32x8 z8 = {0, 0, 0, 0, 0, 0, 0, 0};
  __builtin_amdgcn_tensor_load_to_lds(g0, g1, z4, z4, z8, 0);
}
#endif

// ---------------------------------------------------------------------------
// fp32 [K][N] -> bf16 [N][K] tiled transpose (weights prep, all dims %32==0)
// ---------------------------------------------------------------------------
__global__ __launch_bounds__(256) void k_transpose_f32_bf16(
    const float* __restrict__ in, unsigned short* __restrict__ out, int K, int N) {
  __shared__ float tile[32][33];
  const int n0 = blockIdx.x * 32, k0 = blockIdx.y * 32;
  const int tx = threadIdx.x, ty = threadIdx.y;
  for (int j = 0; j < 32; j += 8)
    tile[ty + j][tx] = in[(long)(k0 + ty + j) * N + (n0 + tx)];
  __syncthreads();
  for (int j = 0; j < 32; j += 8)
    out[(long)(n0 + ty + j) * K + (k0 + tx)] = f2bf(tile[tx][ty + j]);
}

// ---------------------------------------------------------------------------
// Embedding gather -> bf16 activations [B*T][512]
// ---------------------------------------------------------------------------
__global__ __launch_bounds__(256) void k_embed_gather(
    const int* __restrict__ tok, const float* __restrict__ emb,
    unsigned short* __restrict__ out) {
  const int idx = blockIdx.x * 256 + threadIdx.x;   // grid sized exactly
  const int m = idx >> 9, k = idx & 511;
  out[idx] = f2bf(emb[(long)tok[m] * 512 + k]);
}

__global__ __launch_bounds__(256) void k_zero_state(
    unsigned short* __restrict__ h0, unsigned short* __restrict__ h1,
    float* __restrict__ c0, float* __restrict__ c1) {
  const int i = blockIdx.x * 256 + threadIdx.x;
  if (i < 64 * 512) { h0[i] = 0; h1[i] = 0; c0[i] = 0.f; c1[i] = 0.f; }
}

// ---------------------------------------------------------------------------
// Tiled WMMA GEMM: C[M,N] = A[M,K](bf16,row) @ Bt[N,K](bf16,row)^T
// 128x128 macro tile, 8 waves (2x4), each wave 4x2 16x16 WMMA tiles, K step 32.
// Tile staging: TDM double-buffered (wave0 issues, TENSORcnt pipelined) with
// hardware LDS padding; fallback = cooperative b128 loads.
// mode 1: scatter-store bf16 Xz at [(t*64+b)*2048 + n]  (m = b*256 + t)
// mode 2: fp32 store C[m*N+n] + bias[n], guarded by m < M
// ---------------------------------------------------------------------------
__global__ __launch_bounds__(256) void k_gemm_bf16(
    const unsigned short* __restrict__ A, int lda, int M, int K,
    const unsigned short* __restrict__ Bt, int N,
    void* __restrict__ Cout, int mode, const float* __restrict__ bias) {
  __shared__ __attribute__((aligned(128))) unsigned short As[2][128][40];
  __shared__ __attribute__((aligned(128))) unsigned short Bs[2][128][40];
  const int tid  = threadIdx.x;
  const int lane = tid & 31;
  const int w    = tid >> 5;
  const int wm   = w & 1;   // 2 M-wave groups of 64 rows
  const int wn   = w >> 1;  // 4 N-wave groups of 32 cols
  const int m0   = blockIdx.y * 128;
  const int n0   = blockIdx.x * 128;

  v8f acc[4][2];
  const v8f vzero = {0.f, 0.f, 0.f, 0.f, 0.f, 0.f, 0.f, 0.f};
  for (int i = 0; i < 4; ++i)
    for (int j = 0; j < 2; ++j) acc[i][j] = vzero;

  const int lr    = lane & 15;
  const int khalf = lane >> 4;

#if USE_TDM
  const unsigned aLds[2] = {(unsigned)(unsigned long long)&As[0][0][0],
                            (unsigned)(unsigned long long)&As[1][0][0]};
  const unsigned bLds[2] = {(unsigned)(unsigned long long)&Bs[0][0][0],
                            (unsigned)(unsigned long long)&Bs[1][0][0]};
  if (tid < 32) {  // wave 0 drives the TDM pipeline (EXEC-independent issue)
    tdm_load_tile_128x32(A, (long)m0 * lda, lda, M - m0, aLds[0]);
    tdm_load_tile_128x32(Bt, (long)n0 * K, K, 128, bLds[0]);
  }
#else
  const int lrow = tid >> 1;          // 0..127
  const int lcol = (tid & 1) * 16;    // 0 / 16
#endif

  int cur = 0;
  for (int k0 = 0; k0 < K; k0 += 32) {
    const bool more = (k0 + 32 < K);
#if USE_TDM
    if (tid < 32) {
      if (more) {  // prefetch next tiles into the other buffer
        tdm_load_tile_128x32(A, (long)m0 * lda + (k0 + 32), lda, M - m0,
                             aLds[cur ^ 1]);
        tdm_load_tile_128x32(Bt, (long)n0 * K + (k0 + 32), K, 128,
                             bLds[cur ^ 1]);
      }
      if (more) __builtin_amdgcn_s_wait_tensorcnt(2);  // current pair done
      else      __builtin_amdgcn_s_wait_tensorcnt(0);
    }
#else
    {
      const int gm = m0 + lrow;
      U4 a0 = {0, 0, 0, 0}, a1 = {0, 0, 0, 0};
      if (gm < M) {
        const U4* p = (const U4*)&A[(long)gm * lda + k0 + lcol];
        a0 = p[0]; a1 = p[1];
      }
      *(U4*)&As[cur][lrow][lcol]     = a0;
      *(U4*)&As[cur][lrow][lcol + 8] = a1;
      const U4* q = (const U4*)&Bt[(long)(n0 + lrow) * K + k0 + lcol];
      *(U4*)&Bs[cur][lrow][lcol]     = q[0];
      *(U4*)&Bs[cur][lrow][lcol + 8] = q[1];
    }
#endif
    __syncthreads();

    Fr16 af[4], bfr[2];
    for (int mt = 0; mt < 4; ++mt) {
      const int row = wm * 64 + mt * 16 + lr;
      af[mt].u[0] = *(const U4*)&As[cur][row][khalf * 8];
      af[mt].u[1] = *(const U4*)&As[cur][row][16 + khalf * 8];
    }
    for (int nt = 0; nt < 2; ++nt) {
      const int nrow = wn * 32 + nt * 16 + lr;
      bfr[nt].u[0] = *(const U4*)&Bs[cur][nrow][khalf * 8];
      bfr[nt].u[1] = *(const U4*)&Bs[cur][nrow][16 + khalf * 8];
    }
    for (int mt = 0; mt < 4; ++mt)
      for (int nt = 0; nt < 2; ++nt)
        acc[mt][nt] = wmma_bf16(af[mt], bfr[nt], acc[mt][nt]);
    __syncthreads();   // buffer reusable before its next TDM issue
    cur ^= 1;
  }

  for (int mt = 0; mt < 4; ++mt)
    for (int nt = 0; nt < 2; ++nt) {
      const int gn      = n0 + wn * 32 + nt * 16 + lr;
      const int gm_base = m0 + wm * 64 + mt * 16 + (khalf << 3);
      for (int r = 0; r < 8; ++r) {
        const int gm  = gm_base + r;
        const float v = acc[mt][nt][r];
        if (mode == 1) {
          const int b = gm >> 8, t = gm & 255;      // m = b*256 + t
          ((unsigned short*)Cout)[((long)t * 64 + b) * 2048 + gn] = f2bf(v);
        } else if (gm < M) {
          ((float*)Cout)[(long)gm * N + gn] = v + (bias ? bias[gn] : 0.f);
        }
      }
    }
}

// ---------------------------------------------------------------------------
// One LSTM timestep, both directions. 64 WGs: blk>>5 = dir, blk&31 = 16-wide
// h-column slice. Z = h_in @ U (WMMA, K=512) + Xz[t] + bias; gates via LDS.
// h double-buffered across launches; c slice-exclusive per WG. U and h live
// in L2 (~4.25MB total << 192MB); prefetch next step's Xz across launches.
// ---------------------------------------------------------------------------
__global__ __launch_bounds__(256) void k_lstm_step(
    const unsigned short* __restrict__ xzF, const unsigned short* __restrict__ xzB,
    const unsigned short* __restrict__ UtF, const unsigned short* __restrict__ UtB,
    const float* __restrict__ biasF, const float* __restrict__ biasB,
    const unsigned short* __restrict__ hFin, unsigned short* __restrict__ hFout,
    const unsigned short* __restrict__ hBin, unsigned short* __restrict__ hBout,
    float* __restrict__ cF, float* __restrict__ cB,
    unsigned short* __restrict__ outSeq, int s) {
  __shared__ float Zs[4][64][16];
  const int blk    = blockIdx.x;
  const int dir    = blk >> 5;
  const int hc0    = (blk & 31) * 16;
  const int t      = dir ? (255 - s) : s;
  const unsigned short* xz  = dir ? xzB : xzF;
  const unsigned short* Ut  = dir ? UtB : UtF;
  const float* bias         = dir ? biasB : biasF;
  const unsigned short* hin = dir ? hBin : hFin;
  unsigned short* hout      = dir ? hBout : hFout;
  float* c                  = dir ? cB : cF;

  const int tid   = threadIdx.x;
  const int lane  = tid & 31;
  const int w     = tid >> 5;
  const int mt    = w & 3;    // 4 row tiles cover B=64
  const int gp    = w >> 2;   // gate pair {2gp, 2gp+1}
  const int lr    = lane & 15;
  const int khalf = lane >> 4;

  const v8f vzero = {0.f, 0.f, 0.f, 0.f, 0.f, 0.f, 0.f, 0.f};
  v8f acc0 = vzero, acc1 = vzero;

  const int n0 = (2 * gp) * 512 + hc0 + lr;
  const int n1 = n0 + 512;
  for (int k0 = 0; k0 < 512; k0 += 32) {
    Fr16 af, b0, b1;
    const unsigned short* pa = &hin[(mt * 16 + lr) * 512 + k0];
    af.u[0] = *(const U4*)(pa + khalf * 8);
    af.u[1] = *(const U4*)(pa + 16 + khalf * 8);
    const U4* pb0 = (const U4*)&Ut[(long)n0 * 512 + k0 + khalf * 8];
    b0.u[0] = pb0[0]; b0.u[1] = pb0[2];
    const U4* pb1 = (const U4*)&Ut[(long)n1 * 512 + k0 + khalf * 8];
    b1.u[0] = pb1[0]; b1.u[1] = pb1[2];
    acc0 = wmma_bf16(af, b0, acc0);
    acc1 = wmma_bf16(af, b1, acc1);
  }

  for (int half = 0; half < 2; ++half) {
    const v8f a = half ? acc1 : acc0;
    const int g = 2 * gp + half;
    const int gc = g * 512 + hc0 + lr;
    for (int r = 0; r < 8; ++r) {
      const int b = mt * 16 + 8 * khalf + r;   // batch row
      Zs[g][b][lr] = a[r] + bf2f(xz[((long)t * 64 + b) * 2048 + gc]) + bias[gc];
    }
  }
  // warm L2 for the next step's Xz slice (persists across step launches)
  if (s < 255) {
    const int tn = dir ? (254 - s) : (s + 1);
    __builtin_prefetch(&xz[((long)tn * 64 + (tid >> 2)) * 2048 + hc0], 0, 1);
  }
  __syncthreads();

  for (int j = 0; j < 4; ++j) {
    const int e   = tid + 256 * j;   // 0..1023 = 64 rows x 16 cols
    const int b   = e >> 4;
    const int col = e & 15;
    const float zi = Zs[0][b][col], zf = Zs[1][b][col];
    const float zg = Zs[2][b][col], zo = Zs[3][b][col];
    const int ci = b * 512 + hc0 + col;
    const float cv = sigf(zf) * c[ci] + sigf(zi) * tanhf(zg);
    const float hv = sigf(zo) * tanhf(cv);
    c[ci]    = cv;
    hout[ci] = f2bf(hv);
    outSeq[((long)b * 256 + t) * 1024 + dir * 512 + hc0 + col] = f2bf(hv);
  }
}

// ---------------------------------------------------------------------------
// Host orchestration (graph-capturable: launches only)
// ---------------------------------------------------------------------------
extern "C" void kernel_launch(void* const* d_in, const int* in_sizes, int n_in,
                              void* d_out, int out_size, void* d_ws, size_t ws_size,
                              hipStream_t stream) {
  const int*   enc_tok = (const int*)d_in[0];
  const int*   dec_tok = (const int*)d_in[1];
  const float* emb     = (const float*)d_in[2];
  const float* W_in    = (const float*)d_in[3];  // [2][2][512][2048]
  const float* W_hid   = (const float*)d_in[4];  // [2][2][1024][2048]
  const float* U_all   = (const float*)d_in[5];  // [4][2][512][2048]
  const float* b_all   = (const float*)d_in[6];  // [4][2][2048]
  const float* dense_W = (const float*)d_in[7];  // [1024][32000]
  const float* dense_b = (const float*)d_in[8];  // [32000]

  char* p = (char*)d_ws;
  auto carve = [&](size_t elems, size_t esz) -> void* {
    void* r = (void*)p;
    p += (elems * esz + 255) & ~(size_t)255;
    return r;
  };

  const int Kl[4] = {512, 1024, 512, 1024};
  unsigned short* Wt[4][2];
  for (int l = 0; l < 4; ++l)
    for (int d = 0; d < 2; ++d)
      Wt[l][d] = (unsigned short*)carve((size_t)2048 * Kl[l], 2);
  unsigned short* Ut[4][2];
  for (int l = 0; l < 4; ++l)
    for (int d = 0; d < 2; ++d)
      Ut[l][d] = (unsigned short*)carve((size_t)2048 * 512, 2);
  unsigned short* dWt  = (unsigned short*)carve((size_t)32000 * 1024, 2);
  unsigned short* Xenc = (unsigned short*)carve((size_t)16384 * 512, 2);
  unsigned short* Xdec = (unsigned short*)carve((size_t)16384 * 512, 2);
  unsigned short* bufA = (unsigned short*)carve((size_t)16384 * 1024, 2);
  unsigned short* bufB = (unsigned short*)carve((size_t)16384 * 1024, 2);
  unsigned short* xzF  = (unsigned short*)carve((size_t)256 * 64 * 2048, 2);
  unsigned short* xzB  = (unsigned short*)carve((size_t)256 * 64 * 2048, 2);
  unsigned short* hF[2], *hB[2];
  for (int i = 0; i < 2; ++i) hF[i] = (unsigned short*)carve(64 * 512, 2);
  for (int i = 0; i < 2; ++i) hB[i] = (unsigned short*)carve(64 * 512, 2);
  float* cF = (float*)carve(64 * 512, 4);
  float* cB = (float*)carve(64 * 512, 4);

  // --- weight prep: fp32 [K][N] -> bf16 [N][K] ---
  for (int l = 0; l < 4; ++l)
    for (int d = 0; d < 2; ++d) {
      const float* src = (l == 0) ? W_in  + (size_t)d * 512 * 2048
                       : (l == 1) ? W_hid + (size_t)d * 1024 * 2048
                       : (l == 2) ? W_in  + (size_t)(2 + d) * 512 * 2048
                                  : W_hid + (size_t)(2 + d) * 1024 * 2048;
      k_transpose_f32_bf16<<<dim3(2048 / 32, Kl[l] / 32), dim3(32, 8), 0, stream>>>(
          src, Wt[l][d], Kl[l], 2048);
      k_transpose_f32_bf16<<<dim3(2048 / 32, 512 / 32), dim3(32, 8), 0, stream>>>(
          U_all + (size_t)(l * 2 + d) * 512 * 2048, Ut[l][d], 512, 2048);
    }
  k_transpose_f32_bf16<<<dim3(32000 / 32, 1024 / 32), dim3(32, 8), 0, stream>>>(
      dense_W, dWt, 1024, 32000);

  // --- embeddings, states ---
  k_embed_gather<<<32768, 256, 0, stream>>>(enc_tok, emb, Xenc);
  k_embed_gather<<<32768, 256, 0, stream>>>(dec_tok, emb, Xdec);
  k_zero_state<<<128, 256, 0, stream>>>(hF[0], hB[0], cF, cB);

  // --- 4 BiLSTM layers; states chain through persistent h/c buffers ---
  const unsigned short* Xl[4] = {Xenc, bufA, Xdec, bufA};
  unsigned short*       Ol[4] = {bufA, bufB, bufA, bufB};
  for (int l = 0; l < 4; ++l) {
    k_gemm_bf16<<<dim3(16, 128), 256, 0, stream>>>(
        Xl[l], Kl[l], 16384, Kl[l], Wt[l][0], 2048, xzF, 1, nullptr);
    k_gemm_bf16<<<dim3(16, 128), 256, 0, stream>>>(
        Xl[l], Kl[l], 16384, Kl[l], Wt[l][1], 2048, xzB, 1, nullptr);
    for (int s = 0; s < 256; ++s) {
      const int i = s & 1, o = i ^ 1;   // ends (s=255) writing slot 0
      k_lstm_step<<<64, 256, 0, stream>>>(
          xzF, xzB, Ut[l][0], Ut[l][1],
          b_all + (size_t)(l * 2) * 2048, b_all + (size_t)(l * 2 + 1) * 2048,
          hF[i], hF[o], hB[i], hB[o], cF, cB, Ol[l], s);
    }
  }

  // --- dense head on last decoder timestep (lda trick: row b at (b*256+255)) ---
  k_gemm_bf16<<<dim3(250, 1), 256, 0, stream>>>(
      bufB + (size_t)255 * 1024, 256 * 1024, 64, 1024, dWt, 32000,
      d_out, 2, dense_b);
}
